// MLP_53807350284779
// MI455X (gfx1250) — matrix operand order repaired and checked by reading, not compiled, not run on hardware
//
#include <hip/hip_runtime.h>
#include <hip/hip_bf16.h>

// ---------------------------------------------------------------------------
// GraphSAGE (3 layers) + global mean pool + BN/tanh MLP head for MI455X.
//
// Roofline: edge gather/scatter traffic ~1.2 GB @ 23.3 TB/s dominates (~53us);
// GEMMs (6x [50000x64]x[64x64], 2.5 GFLOP) run through v_wmma_f32_16x16x32_bf16
// with fp32 accumulation; weights LDS-staged transposed as bf16; B fragments
// are hoisted to registers by the compiler and A rows for the next grid-stride
// tile are prefetched with global_prefetch_b8.
// ---------------------------------------------------------------------------

#define Nn 50000
#define Ee 800000
#define Dd 64
#define Gg 128

typedef __attribute__((ext_vector_type(16))) __bf16 v16bf;
typedef __attribute__((ext_vector_type(8)))  float  v8f;

// ---------------------------------------------------------------------------
// Edge scatter: agg[dst] += h[src], deg[dst] += 1.
// One thread per (edge, 4-float chunk): 16 threads/edge, float4 gather,
// native f32 global atomics (global_atomic_add_f32) for the scatter.
// ---------------------------------------------------------------------------
__global__ __launch_bounds__(256) void scatter_kernel(
    const float* __restrict__ hin, const int* __restrict__ ei,
    float* __restrict__ agg, float* __restrict__ deg)
{
    int tid = blockIdx.x * 256 + threadIdx.x;       // E*16 = 12.8M threads
    int e = tid >> 4, q = tid & 15;
    if (e >= Ee) return;
    unsigned src = (unsigned)ei[e];
    unsigned dst = (unsigned)ei[Ee + e];
    const float4 v = *(const float4*)(hin + ((size_t)src << 6) + (q << 2));
    float* p = agg + ((size_t)dst << 6) + (q << 2);
    unsafeAtomicAdd(p + 0, v.x);
    unsafeAtomicAdd(p + 1, v.y);
    unsafeAtomicAdd(p + 2, v.z);
    unsafeAtomicAdd(p + 3, v.w);
    if (q == 0) unsafeAtomicAdd(deg + dst, 1.0f);
}

// ---------------------------------------------------------------------------
// Fused SAGE linear: hout = (agg/max(deg,1)) @ wl + bl + hin @ wr
// One wave = one 16-row tile (N = 3125 exact tiles). Both products accumulate
// into the same C fragments via v_wmma_f32_16x16x32_bf16 (K=64 -> 2 steps,
// 4 column tiles, 2 matrices = 16 WMMA per tile).
// ---------------------------------------------------------------------------
__global__ __launch_bounds__(256) void sage_gemm_kernel(
    const float* __restrict__ hin, const float* __restrict__ agg,
    const float* __restrict__ deg,
    const float* __restrict__ wl, const float* __restrict__ bias,
    const float* __restrict__ wr,
    float* __restrict__ hout)
{
    // Transposed bf16 weights in LDS: sW[n*64 + k] = W[k][n] -> contiguous-K
    // B-fragment loads (32 bytes per lane).
    __shared__ __bf16 sWl[64 * 64];
    __shared__ __bf16 sWr[64 * 64];
    __shared__ float  sB[64];
    for (int idx = threadIdx.x; idx < 4096; idx += 256) {
        int k = idx >> 6, n = idx & 63;
        sWl[n * 64 + k] = (__bf16)wl[idx];
        sWr[n * 64 + k] = (__bf16)wr[idx];
    }
    if (threadIdx.x < 64) sB[threadIdx.x] = bias[threadIdx.x];
    __syncthreads();

    const int lane = threadIdx.x & 31;
    const int wave = threadIdx.x >> 5;
    const int m    = lane & 15;   // row-in-tile (A) / column-in-tile (B,C)
    const int half = lane >> 4;
    const int NT   = Nn / 16;     // 3125 tiles
    const int step = gridDim.x * 8;

    for (int tile = blockIdx.x * 8 + wave; tile < NT; tile += step) {
        const int row = tile * 16 + m;
        const float invd = 1.0f / fmaxf(deg[row], 1.0f);
        // Fold the half*8 K-offset into the base so all chunk loads use
        // compile-time immediate offsets (0, +16, +32, +48 floats).
        const float* pA = agg + ((size_t)row << 6) + half * 8;
        const float* pH = hin + ((size_t)row << 6) + half * 8;

        // Prefetch next grid-stride tile's rows (one 256B row per lane).
        const int nrow = row + step * 16;
        if (nrow < Nn) {
            __builtin_prefetch(agg + ((size_t)nrow << 6), 0, 1);
            __builtin_prefetch(hin + ((size_t)nrow << 6), 0, 1);
        }

        // C init: every element of column-tile t gets bias[t*16 + n].
        v8f c[4];
        #pragma unroll
        for (int t = 0; t < 4; ++t) {
            float bv = sB[t * 16 + m];
            #pragma unroll
            for (int r = 0; r < 8; ++r) c[t][r] = bv;
        }

        #pragma unroll
        for (int s = 0; s < 2; ++s) {           // K steps of 32
            // A fragment (16-bit 16x32 layout): lane m+16*half holds
            // K = {s*32 + half*8 + 0..7} and {s*32 + 16 + half*8 + 0..7}.
            const int bk = s * 32;
            v16bf aM, aH;
            #pragma unroll
            for (int e2 = 0; e2 < 8; ++e2) {
                aM[e2]     = (__bf16)(pA[bk + e2] * invd);
                aM[8 + e2] = (__bf16)(pA[bk + 16 + e2] * invd);
                aH[e2]     = (__bf16)(pH[bk + e2]);
                aH[8 + e2] = (__bf16)(pH[bk + 16 + e2]);
            }
            #pragma unroll
            for (int t = 0; t < 4; ++t) {       // 16-column tiles
                // B fragment: lane n+16*half holds W[s*32+half*8 .. +7][n] and W[s*32+16+half*8 .. +7][n].
                const int n  = t * 16 + m;
                const int kk = s * 32 + half * 8;
                v16bf bL, bR;
                #pragma unroll
                for (int e2 = 0; e2 < 16; ++e2) {
                    bL[e2] = sWl[n * 64 + kk + e2 + (e2 & 8)];
                    bR[e2] = sWr[n * 64 + kk + e2 + (e2 & 8)];
                }
                c[t] = __builtin_amdgcn_wmma_f32_16x16x32_bf16(
                    false, aM, false, bL, (short)0, c[t], false, false);
                c[t] = __builtin_amdgcn_wmma_f32_16x16x32_bf16(
                    false, aH, false, bR, (short)0, c[t], false, false);
            }
        }

        // C layout: lane -> N = lane%16, VGPR r -> M = r + 8*(lane/16).
        // Single base pointer; all 32 stores use immediate offsets.
        float* pout = hout + (((size_t)tile * 16 + 8 * half) << 6) + m;
        #pragma unroll
        for (int t = 0; t < 4; ++t)
            #pragma unroll
            for (int r = 0; r < 8; ++r)
                pout[r * Dd + t * 16] = c[t][r];
    }
}

// ---------------------------------------------------------------------------
// Global mean pool: psum[batch[i]] += h[i], pcnt[batch[i]] += 1.
// ---------------------------------------------------------------------------
__global__ __launch_bounds__(256) void pool_kernel(
    const float* __restrict__ h, const int* __restrict__ batch,
    float* __restrict__ psum, float* __restrict__ pcnt)
{
    int tid = blockIdx.x * 256 + threadIdx.x;       // N*16 threads
    int i = tid >> 4, q = tid & 15;
    if (i >= Nn) return;
    unsigned g = (unsigned)batch[i];
    const float4 v = *(const float4*)(h + ((size_t)i << 6) + (q << 2));
    float* p = psum + ((size_t)g << 6) + (q << 2);
    unsafeAtomicAdd(p + 0, v.x);
    unsafeAtomicAdd(p + 1, v.y);
    unsafeAtomicAdd(p + 2, v.z);
    unsafeAtomicAdd(p + 3, v.w);
    if (q == 0) unsafeAtomicAdd(pcnt + g, 1.0f);
}

__global__ void pool_fin_kernel(const float* __restrict__ psum,
                                const float* __restrict__ pcnt,
                                float* __restrict__ c)
{
    int i = blockIdx.x * blockDim.x + threadIdx.x;
    if (i < Gg * Dd) c[i] = psum[i] / fmaxf(pcnt[i >> 6], 1.0f);
}

// ---------------------------------------------------------------------------
// Fused linear + batch-norm (batch stats over G=128 rows) + tanh.
// One block per output column j; thread g = row. LDS tree reduction.
// ---------------------------------------------------------------------------
__global__ __launch_bounds__(128) void lin_bn_tanh_kernel(
    const float* __restrict__ in, int K,
    const float* __restrict__ W, const float* __restrict__ b,
    const float* __restrict__ gamma, const float* __restrict__ beta,
    float* __restrict__ out, int M)
{
    const int j = blockIdx.x;
    const int g = threadIdx.x;  // 0..127 (G rows)
    float s = b[j];
    for (int k = 0; k < K; ++k) s += in[g * K + k] * W[k * M + j];

    __shared__ float red[128];
    red[g] = s;
    __syncthreads();
    for (int off = 64; off > 0; off >>= 1) {
        if (g < off) red[g] += red[g + off];
        __syncthreads();
    }
    const float mean = red[0] * (1.0f / 128.0f);
    __syncthreads();
    red[g] = s * s;
    __syncthreads();
    for (int off = 64; off > 0; off >>= 1) {
        if (g < off) red[g] += red[g + off];
        __syncthreads();
    }
    const float var = red[0] * (1.0f / 128.0f) - mean * mean;

    const float y = (s - mean) * rsqrtf(var + 1e-5f) * gamma[j] + beta[j];
    out[g * M + j] = tanhf(y);
}

__global__ __launch_bounds__(128) void lin_kernel(
    const float* __restrict__ in, int K,
    const float* __restrict__ W, const float* __restrict__ b,
    float* __restrict__ out, int M)
{
    const int j = blockIdx.x;
    const int g = threadIdx.x;
    float s = b[j];
    for (int k = 0; k < K; ++k) s += in[g * K + k] * W[k * M + j];
    out[g * M + j] = s;
}

// ---------------------------------------------------------------------------

extern "C" void kernel_launch(void* const* d_in, const int* in_sizes, int n_in,
                              void* d_out, int out_size, void* d_ws, size_t ws_size,
                              hipStream_t stream)
{
    const float* x     = (const float*)d_in[0];
    const int*   ei    = (const int*)  d_in[1];
    const int*   batch = (const int*)  d_in[2];
    const float* w1l = (const float*)d_in[3];
    const float* b1l = (const float*)d_in[4];
    const float* w1r = (const float*)d_in[5];
    const float* w2l = (const float*)d_in[6];
    const float* b2l = (const float*)d_in[7];
    const float* w2r = (const float*)d_in[8];
    const float* w3l = (const float*)d_in[9];
    const float* b3l = (const float*)d_in[10];
    const float* w3r = (const float*)d_in[11];
    const float* lin1_w = (const float*)d_in[12];
    const float* lin1_b = (const float*)d_in[13];
    const float* g1  = (const float*)d_in[14];
    const float* be1 = (const float*)d_in[15];
    const float* lin2_w = (const float*)d_in[16];
    const float* lin2_b = (const float*)d_in[17];
    const float* g2  = (const float*)d_in[18];
    const float* be2 = (const float*)d_in[19];
    const float* lin3_w = (const float*)d_in[20];
    const float* lin3_b = (const float*)d_in[21];
    const float* g3  = (const float*)d_in[22];
    const float* be3 = (const float*)d_in[23];
    const float* lin4_w = (const float*)d_in[24];
    const float* lin4_b = (const float*)d_in[25];

    // Workspace layout (floats), ~38.9 MB total.
    float* ws    = (float*)d_ws;
    float* agg   = ws;                       // N*D = 3,200,000
    float* deg   = agg + (size_t)Nn * Dd;    // N   = 50,000 (contiguous w/ agg)
    float* h1    = deg + Nn;                 // N*D
    float* h2    = h1 + (size_t)Nn * Dd;     // N*D
    float* psum  = h2 + (size_t)Nn * Dd;     // G*D = 8192
    float* pcnt  = psum + (size_t)Gg * Dd;   // G   = 128
    float* cpool = pcnt + Gg;                // G*D
    float* o1    = cpool + (size_t)Gg * Dd;  // G*200
    float* o2    = o1 + Gg * 200;            // G*100
    float* o3    = o2 + Gg * 100;            // G*100

    const int scatterBlocks = (Ee * 16) / 256;   // 50000
    const int poolBlocks    = (Nn * 16) / 256;   // 3125

    auto sage = [&](const float* hin, const float* wl, const float* bl,
                    const float* wr, float* hout) {
        hipMemsetAsync(agg, 0, ((size_t)Nn * Dd + Nn) * sizeof(float), stream);
        scatter_kernel<<<scatterBlocks, 256, 0, stream>>>(hin, ei, agg, deg);
        sage_gemm_kernel<<<160, 256, 0, stream>>>(hin, agg, deg, wl, bl, wr, hout);
    };

    sage(x,  w1l, b1l, w1r, h1);
    sage(h1, w2l, b2l, w2r, h2);
    sage(h2, w3l, b3l, w3r, h1);

    hipMemsetAsync(psum, 0, ((size_t)Gg * Dd + Gg) * sizeof(float), stream);
    pool_kernel<<<poolBlocks, 256, 0, stream>>>(h1, batch, psum, pcnt);
    pool_fin_kernel<<<(Gg * Dd + 255) / 256, 256, 0, stream>>>(psum, pcnt, cpool);

    lin_bn_tanh_kernel<<<200, 128, 0, stream>>>(cpool, 64,  lin1_w, lin1_b, g1, be1, o1, 200);
    lin_bn_tanh_kernel<<<100, 128, 0, stream>>>(o1,    200, lin2_w, lin2_b, g2, be2, o2, 100);
    lin_bn_tanh_kernel<<<100, 128, 0, stream>>>(o2,    100, lin3_w, lin3_b, g3, be3, o3, 100);
    lin_kernel<<<80, 128, 0, stream>>>(o3, 100, lin4_w, lin4_b, (float*)d_out, 80);
}
